// GAT_59545426591792
// MI455X (gfx1250) — matrix operand order repaired and checked by reading, not compiled, not run on hardware
//
#include <hip/hip_runtime.h>
#include <math.h>

#define NNODES 50000
#define FDIM   128
#define CDIM   16
#define NEG_SLOPE 0.2f
#define EPS_DEN 1e-16f

typedef __attribute__((ext_vector_type(16))) __bf16 v16bf;
typedef __attribute__((ext_vector_type(8)))  float  v8f;
typedef __attribute__((ext_vector_type(8)))  unsigned short v8u16;
typedef __attribute__((ext_vector_type(16))) unsigned short v16u16;

__device__ __forceinline__ unsigned short f32_to_bf16(float f) {
  unsigned int u = __float_as_uint(f);
  u += 0x7fffu + ((u >> 16) & 1u);          // round-to-nearest-even
  return (unsigned short)(u >> 16);
}

// float atomic max via signed/unsigned integer ordering trick (works for mixed signs)
__device__ __forceinline__ void atomicMaxF32(float* addr, float val) {
  if (val >= 0.f) atomicMax((int*)addr, __float_as_int(val));
  else            atomicMin((unsigned int*)addr, __float_as_uint(val));
}

// ---------------- conversion kernels ----------------
__global__ void cvt_bf16_kernel(const float* __restrict__ in,
                                unsigned short* __restrict__ out, int n) {
  int i = blockIdx.x * blockDim.x + threadIdx.x;
  if (i < n) out[i] = f32_to_bf16(in[i]);
}

// in: [K][Ncols] row-major f32  ->  out: [Ncols][K] row-major bf16 (transposed)
__global__ void cvtT_bf16_kernel(const float* __restrict__ in,
                                 unsigned short* __restrict__ out, int K, int Ncols) {
  int i = blockIdx.x * blockDim.x + threadIdx.x;
  if (i < K * Ncols) {
    int k = i / Ncols, n = i % Ncols;
    out[n * K + k] = f32_to_bf16(in[i]);
  }
}

__global__ void fill_kernel(float* __restrict__ p, int n, float v) {
  int i = blockIdx.x * blockDim.x + threadIdx.x;
  if (i < n) p[i] = v;
}

// ---------------- WMMA GEMM 1: h1[50000x128] = x[50000x128] @ W1[128x128] ----------------
// xb: bf16 row-major [M][K]; w1t: bf16 [N][K] (W1 transposed). 8 waves/block,
// wave w -> 16x16 tile at (blockIdx.x*16, w*16). K loop = 4 steps of 32.
__global__ void gemm1_kernel(const unsigned short* __restrict__ xb,
                             const unsigned short* __restrict__ w1t,
                             float* __restrict__ h1) {
  const int wave = threadIdx.x >> 5;
  const int lane = threadIdx.x & 31;
  const int hlf  = lane >> 4;          // 0: lanes 0-15, 1: lanes 16-31
  const int l16  = lane & 15;
  const int rowBase = blockIdx.x * 16;
  const int colBase = wave * 16;

  const unsigned short* arow = xb  + (size_t)(rowBase + l16) * FDIM;
  const unsigned short* brow = w1t + (size_t)(colBase + l16) * FDIM;

  v8f acc = {};
#pragma unroll
  for (int k0 = 0; k0 < FDIM; k0 += 32) {
    // A 16x32 bf16 fragment: lane half 0 -> K k0+0..7 / k0+16..23 ; half 1 -> +8 / +24
    const unsigned short* ap = arow + k0 + hlf * 8;
    v8u16 a0 = *(const v8u16*)(ap);
    v8u16 a1 = *(const v8u16*)(ap + 16);
    v16u16 av = __builtin_shufflevector(a0, a1, 0,1,2,3,4,5,6,7,8,9,10,11,12,13,14,15);
    // B 32x16 bf16 fragment: half 0 -> K k0+0..7 / k0+16..23 ; half 1 -> +8 / +24
    const unsigned short* bp = brow + k0 + hlf * 8;
    v8u16 b0 = *(const v8u16*)(bp);
    v8u16 b1 = *(const v8u16*)(bp + 16);
    v16u16 bv = __builtin_shufflevector(b0, b1, 0,1,2,3,4,5,6,7,8,9,10,11,12,13,14,15);

    acc = __builtin_amdgcn_wmma_f32_16x16x32_bf16(
        false, __builtin_bit_cast(v16bf, av),
        false, __builtin_bit_cast(v16bf, bv),
        (short)0, acc, false, false);
  }
  // C layout: VGPR i -> row rowBase + i + hlf*8, col colBase + l16
#pragma unroll
  for (int i = 0; i < 8; ++i)
    h1[(size_t)(rowBase + i + hlf * 8) * FDIM + colBase + l16] = acc[i];
}

// ---------------- WMMA GEMM 2: h2[50000x16] = h1r[50000x128] @ W2[128x16] ----------------
__global__ void gemm2_kernel(const unsigned short* __restrict__ hb,
                             const unsigned short* __restrict__ w2t,
                             float* __restrict__ h2) {
  const int wave = threadIdx.x >> 5;
  const int lane = threadIdx.x & 31;
  const int hlf  = lane >> 4;
  const int l16  = lane & 15;
  const int rowTile = blockIdx.x * 8 + wave;
  if (rowTile >= NNODES / 16) return;
  const int rowBase = rowTile * 16;

  const unsigned short* arow = hb  + (size_t)(rowBase + l16) * FDIM;
  const unsigned short* brow = w2t + (size_t)l16 * FDIM;   // 16 cols

  v8f acc = {};
#pragma unroll
  for (int k0 = 0; k0 < FDIM; k0 += 32) {
    const unsigned short* ap = arow + k0 + hlf * 8;
    v8u16 a0 = *(const v8u16*)(ap);
    v8u16 a1 = *(const v8u16*)(ap + 16);
    v16u16 av = __builtin_shufflevector(a0, a1, 0,1,2,3,4,5,6,7,8,9,10,11,12,13,14,15);
    const unsigned short* bp = brow + k0 + hlf * 8;
    v8u16 b0 = *(const v8u16*)(bp);
    v8u16 b1 = *(const v8u16*)(bp + 16);
    v16u16 bv = __builtin_shufflevector(b0, b1, 0,1,2,3,4,5,6,7,8,9,10,11,12,13,14,15);

    acc = __builtin_amdgcn_wmma_f32_16x16x32_bf16(
        false, __builtin_bit_cast(v16bf, av),
        false, __builtin_bit_cast(v16bf, bv),
        (short)0, acc, false, false);
  }
#pragma unroll
  for (int i = 0; i < 8; ++i)
    h2[(size_t)(rowBase + i + hlf * 8) * CDIM + l16] = acc[i];
}

// ---------------- attention coefficient dots ----------------
__global__ void alpha1_kernel(const float* __restrict__ h1,
                              const float* __restrict__ avs, const float* __restrict__ avd,
                              float* __restrict__ as, float* __restrict__ ad) {
  int wid  = (blockIdx.x * blockDim.x + threadIdx.x) >> 5;
  int lane = threadIdx.x & 31;
  if (wid >= NNODES) return;
  const float* row = h1 + (size_t)wid * FDIM;
  float ss = 0.f, sd = 0.f;
#pragma unroll
  for (int j = lane; j < FDIM; j += 32) {
    float v = row[j];
    ss += v * avs[j];
    sd += v * avd[j];
  }
#pragma unroll
  for (int o = 16; o > 0; o >>= 1) {
    ss += __shfl_xor(ss, o, 32);
    sd += __shfl_xor(sd, o, 32);
  }
  if (lane == 0) { as[wid] = ss; ad[wid] = sd; }
}

__global__ void alpha2_kernel(const float* __restrict__ h2,
                              const float* __restrict__ avs, const float* __restrict__ avd,
                              float* __restrict__ as, float* __restrict__ ad) {
  int n = blockIdx.x * blockDim.x + threadIdx.x;
  if (n >= NNODES) return;
  float ss = 0.f, sd = 0.f;
#pragma unroll
  for (int c = 0; c < CDIM; ++c) {
    float v = h2[(size_t)n * CDIM + c];
    ss += v * avs[c];
    sd += v * avd[c];
  }
  as[n] = ss; ad[n] = sd;
}

// ---------------- edge passes ----------------
__global__ void epass1_kernel(const int* __restrict__ src, const int* __restrict__ dst,
                              int E0, int ET,
                              const float* __restrict__ as, const float* __restrict__ ad,
                              float* __restrict__ ebuf, float* __restrict__ mbuf) {
  int e = blockIdx.x * blockDim.x + threadIdx.x;
  if (e >= ET) return;
  int s, d;
  if (e < E0) { s = src[e]; d = dst[e]; } else { s = e - E0; d = s; }
  float v = as[s] + ad[d];
  v = (v >= 0.f) ? v : NEG_SLOPE * v;
  ebuf[e] = v;
  atomicMaxF32(mbuf + d, v);
}

__global__ void epass2_kernel(const int* __restrict__ src, const int* __restrict__ dst,
                              int E0, int ET,
                              const float* __restrict__ mbuf,
                              float* __restrict__ ebuf, float* __restrict__ dbuf) {
  int e = blockIdx.x * blockDim.x + threadIdx.x;
  if (e >= ET) return;
  int d = (e < E0) ? dst[e] : (e - E0);
  float ex = __expf(ebuf[e] - mbuf[d]);
  ebuf[e] = ex;
  atomicAdd(dbuf + d, ex);
}

// wave per edge, 128 features (float4 per lane), L2-resident atomics
__global__ void eagg1_kernel(const int* __restrict__ src, const int* __restrict__ dst,
                             int E0, int ET,
                             const float* __restrict__ ebuf, const float* __restrict__ dbuf,
                             const float* __restrict__ h1, float* __restrict__ agg) {
  int wid  = (blockIdx.x * blockDim.x + threadIdx.x) >> 5;
  int lane = threadIdx.x & 31;
  if (wid >= ET) return;
  int s, d;
  if (wid < E0) { s = src[wid]; d = dst[wid]; } else { s = wid - E0; d = s; }
  float w = ebuf[wid] / (dbuf[d] + EPS_DEN);
  const float4 hv = ((const float4*)(h1 + (size_t)s * FDIM))[lane];
  float* ap = agg + (size_t)d * FDIM + lane * 4;
  atomicAdd(ap + 0, w * hv.x);
  atomicAdd(ap + 1, w * hv.y);
  atomicAdd(ap + 2, w * hv.z);
  atomicAdd(ap + 3, w * hv.w);
}

// thread per (edge, class): 16 features
__global__ void eagg2_kernel(const int* __restrict__ src, const int* __restrict__ dst,
                             int E0, int ET,
                             const float* __restrict__ ebuf, const float* __restrict__ dbuf,
                             const float* __restrict__ h2, float* __restrict__ agg) {
  long long t = (long long)blockIdx.x * blockDim.x + threadIdx.x;
  int e = (int)(t >> 4);
  int c = (int)(t & 15);
  if (e >= ET) return;
  int s, d;
  if (e < E0) { s = src[e]; d = dst[e]; } else { s = e - E0; d = s; }
  float w = ebuf[e] / (dbuf[d] + EPS_DEN);
  atomicAdd(agg + (size_t)d * CDIM + c, w * h2[(size_t)s * CDIM + c]);
}

// h1r = bf16(relu(agg1 + b1))
__global__ void finalize1_kernel(const float* __restrict__ agg, const float* __restrict__ b1,
                                 unsigned short* __restrict__ hb, int n) {
  int i = blockIdx.x * blockDim.x + threadIdx.x;
  if (i >= n) return;
  float v = agg[i] + b1[i & (FDIM - 1)];
  v = (v > 0.f) ? v : 0.f;
  hb[i] = f32_to_bf16(v);
}

// out = log_softmax(agg2 + b2) row-wise over 16 classes
__global__ void lsm_kernel(const float* __restrict__ agg, const float* __restrict__ b2,
                           float* __restrict__ out) {
  int n = blockIdx.x * blockDim.x + threadIdx.x;
  if (n >= NNODES) return;
  float v[CDIM];
  float m = -INFINITY;
#pragma unroll
  for (int c = 0; c < CDIM; ++c) {
    v[c] = agg[(size_t)n * CDIM + c] + b2[c];
    m = fmaxf(m, v[c]);
  }
  float s = 0.f;
#pragma unroll
  for (int c = 0; c < CDIM; ++c) s += __expf(v[c] - m);
  float ls = __logf(s);
#pragma unroll
  for (int c = 0; c < CDIM; ++c) out[(size_t)n * CDIM + c] = v[c] - m - ls;
}

// ---------------- host launcher ----------------
static inline size_t alignup(size_t x) { return (x + 255) & ~(size_t)255; }

extern "C" void kernel_launch(void* const* d_in, const int* in_sizes, int n_in,
                              void* d_out, int out_size, void* d_ws, size_t ws_size,
                              hipStream_t stream) {
  const float* x     = (const float*)d_in[0];
  const int*   ei    = (const int*)  d_in[1];
  const float* W1    = (const float*)d_in[2];
  const float* as1v  = (const float*)d_in[3];
  const float* ad1v  = (const float*)d_in[4];
  const float* b1    = (const float*)d_in[5];
  const float* W2    = (const float*)d_in[6];
  const float* as2v  = (const float*)d_in[7];
  const float* ad2v  = (const float*)d_in[8];
  const float* b2    = (const float*)d_in[9];
  float* out = (float*)d_out;

  const int E0 = in_sizes[1] / 2;        // original edges
  const int ET = E0 + NNODES;            // + self loops
  const int* srcI = ei;
  const int* dstI = ei + E0;

  // workspace bump allocator
  char* p = (char*)d_ws;
  size_t off = 0;
  auto grab = [&](size_t bytes) { char* r = p + off; off = alignup(off + bytes); return (void*)r; };

  unsigned short* xb   = (unsigned short*)grab((size_t)NNODES * FDIM * 2);
  unsigned short* w1t  = (unsigned short*)grab((size_t)FDIM * FDIM * 2);
  unsigned short* w2t  = (unsigned short*)grab((size_t)CDIM * FDIM * 2);
  float* h1   = (float*)grab((size_t)NNODES * FDIM * 4);
  float* as1  = (float*)grab((size_t)NNODES * 4);
  float* ad1  = (float*)grab((size_t)NNODES * 4);
  float* ebuf = (float*)grab((size_t)ET * 4);
  float* m1   = (float*)grab((size_t)NNODES * 4);
  float* dn1  = (float*)grab((size_t)NNODES * 4);
  float* agg1 = (float*)grab((size_t)NNODES * FDIM * 4);
  unsigned short* h1rb = (unsigned short*)grab((size_t)NNODES * FDIM * 2);
  float* h2   = (float*)grab((size_t)NNODES * CDIM * 4);
  float* as2  = (float*)grab((size_t)NNODES * 4);
  float* ad2  = (float*)grab((size_t)NNODES * 4);
  float* m2   = (float*)grab((size_t)NNODES * 4);
  float* dn2  = (float*)grab((size_t)NNODES * 4);
  float* agg2 = (float*)grab((size_t)NNODES * CDIM * 4);
  (void)ws_size; (void)n_in; (void)out_size;

  const int B = 256;
  const int nxf = NNODES * FDIM;           // 6.4M
  const int ncf = NNODES * CDIM;           // 800K

  // 1. conversions
  cvt_bf16_kernel <<<(nxf + B - 1) / B, B, 0, stream>>>(x, xb, nxf);
  cvtT_bf16_kernel<<<(FDIM * FDIM + B - 1) / B, B, 0, stream>>>(W1, w1t, FDIM, FDIM);
  cvtT_bf16_kernel<<<(FDIM * CDIM + B - 1) / B, B, 0, stream>>>(W2, w2t, FDIM, CDIM);

  // 2. init accumulators (ws is poisoned; must re-init every call)
  fill_kernel<<<(nxf + B - 1) / B, B, 0, stream>>>(agg1, nxf, 0.f);
  fill_kernel<<<(NNODES + B - 1) / B, B, 0, stream>>>(m1, NNODES, -INFINITY);
  fill_kernel<<<(NNODES + B - 1) / B, B, 0, stream>>>(dn1, NNODES, 0.f);
  fill_kernel<<<(ncf + B - 1) / B, B, 0, stream>>>(agg2, ncf, 0.f);
  fill_kernel<<<(NNODES + B - 1) / B, B, 0, stream>>>(m2, NNODES, -INFINITY);
  fill_kernel<<<(NNODES + B - 1) / B, B, 0, stream>>>(dn2, NNODES, 0.f);

  // 3. layer 1: GEMM (WMMA), alphas, edge softmax + aggregation
  gemm1_kernel<<<NNODES / 16, 256, 0, stream>>>(xb, w1t, h1);
  alpha1_kernel<<<(NNODES * 32 + B - 1) / B, B, 0, stream>>>(h1, as1v, ad1v, as1, ad1);
  epass1_kernel<<<(ET + B - 1) / B, B, 0, stream>>>(srcI, dstI, E0, ET, as1, ad1, ebuf, m1);
  epass2_kernel<<<(ET + B - 1) / B, B, 0, stream>>>(srcI, dstI, E0, ET, m1, ebuf, dn1);
  eagg1_kernel<<<((long long)ET * 32 + B - 1) / B, B, 0, stream>>>(srcI, dstI, E0, ET,
                                                                   ebuf, dn1, h1, agg1);
  finalize1_kernel<<<(nxf + B - 1) / B, B, 0, stream>>>(agg1, b1, h1rb, nxf);

  // 4. layer 2
  gemm2_kernel<<<(NNODES / 16 + 7) / 8, 256, 0, stream>>>(h1rb, w2t, h2);
  alpha2_kernel<<<(NNODES + B - 1) / B, B, 0, stream>>>(h2, as2v, ad2v, as2, ad2);
  epass1_kernel<<<(ET + B - 1) / B, B, 0, stream>>>(srcI, dstI, E0, ET, as2, ad2, ebuf, m2);
  epass2_kernel<<<(ET + B - 1) / B, B, 0, stream>>>(srcI, dstI, E0, ET, m2, ebuf, dn2);
  eagg2_kernel<<<((long long)ET * CDIM + B - 1) / B, B, 0, stream>>>(srcI, dstI, E0, ET,
                                                                     ebuf, dn2, h2, agg2);

  // 5. log-softmax output
  lsm_kernel<<<(NNODES + B - 1) / B, B, 0, stream>>>(agg2, b2, out);
}
